// JTF_16801912062604
// MI455X (gfx1250) — matrix operand order repaired and checked by reading, not compiled, not run on hardware
//
#include <hip/hip_runtime.h>

// ---------------------------------------------------------------------------
// CDNA5 / gfx1250: all GEMM work on v_wmma_f32_16x16x32_bf16, activations in
// NHWC bf16 so every stage moves 16B/lane and LDS staging is transpose-free
// (async global->LDS via the gfx1250 builtin when available).
// ---------------------------------------------------------------------------

typedef __attribute__((ext_vector_type(16))) __bf16 v16bf;
typedef __attribute__((ext_vector_type(8)))  __bf16 v8bf;
typedef __attribute__((ext_vector_type(8)))  float  v8f;
typedef __attribute__((ext_vector_type(4)))  float  v4f;

#define HH 256
#define WWD 256
#define CC 64
#define BB 4
#define HW (HH*WWD)
#define KH 72      // K*K*G
#define KHP 80     // padded to 5 tiles of 16

#if defined(__has_builtin)
#if __has_builtin(__builtin_amdgcn_global_load_async_to_lds_b128)
#define USE_ASYNC_LDS 1
#endif
#endif

// Builtin parameter type (from hipcc diagnostic): AS1 pointer to gcc-vector int4.
typedef int v4i_ __attribute__((vector_size(16)));
typedef __attribute__((address_space(1))) v4i_ as1_v4i;
typedef __attribute__((address_space(3))) v4i_ as3_v4i;

__device__ __forceinline__ void async_cp16(const __bf16* src, __bf16* dst) {
#ifdef USE_ASYNC_LDS
    __builtin_amdgcn_global_load_async_to_lds_b128(
        (as1_v4i*)src, (as3_v4i*)dst, 0, 0);
#else
    *(v8bf*)dst = *(const v8bf*)src;
#endif
}
__device__ __forceinline__ void async_wait() {
#ifdef USE_ASYNC_LDS
#if __has_builtin(__builtin_amdgcn_s_wait_asynccnt)
    __builtin_amdgcn_s_wait_asynccnt(0);
#else
    asm volatile("s_wait_asynccnt 0x0" ::: "memory");
#endif
#endif
}

// ----------------------------- layout prep ---------------------------------

// NCHW f32 -> NHWC bf16
__global__ void k_nchw_to_nhwc(const float* __restrict__ s, __bf16* __restrict__ d, int n) {
    int i = blockIdx.x * blockDim.x + threadIdx.x;
    if (i >= n) return;
    int p   = i % HW;
    int ch  = (i / HW) % CC;
    int bat = i / (CC*HW);
    d[((size_t)bat*HW + p)*CC + ch] = (__bf16)s[i];
}

// (L,64,64,3,3) OIHW f32 -> per-layer [tap][co][ci] bf16 (A-fragment rows)
__global__ void k_reorder_w3(const float* __restrict__ w, __bf16* __restrict__ o, int total) {
    int i = blockIdx.x * blockDim.x + threadIdx.x;
    if (i >= total) return;
    int l  = i / (CC*CC*9);
    int r  = i % (CC*CC*9);
    int co = r / (CC*9);
    int r2 = r % (CC*9);
    int ci = r2 / 9;
    int t  = r2 % 9;
    o[l*(9*CC*CC) + (t*CC + co)*CC + ci] = (__bf16)w[i];
}

// head weights (72,64,1,1) f32 -> (80,64) bf16, rows 72..79 zero
__global__ void k_reorder_wh(const float* __restrict__ w, __bf16* __restrict__ o) {
    int i = blockIdx.x * blockDim.x + threadIdx.x;
    if (i >= KHP*CC) return;
    int m = i / CC;
    o[i] = (m < KH) ? (__bf16)w[i] : (__bf16)0.0f;
}

// ----------------------- 3x3 conv via WMMA ---------------------------------
// Block = 8 waves -> 64 out-ch x 32 px of one row. NHWC input: the LDS tile
// [3 rows][34 cols][64 ci] is a set of contiguous 128B pixel rows -> staged
// as 16B/lane copies (async-to-LDS on CDNA5). 18 WMMAs per wave.
template<bool PRELU, bool RESID>
__global__ __launch_bounds__(256)
void k_conv3x3(const __bf16* __restrict__ x, const __bf16* __restrict__ wA,
               const float* __restrict__ bias, const float* __restrict__ alpha,
               __bf16* __restrict__ ybf, float* __restrict__ yf32,
               const float* __restrict__ resid) {
    __shared__ __bf16 xs[3*34*CC];
    const int bx  = blockIdx.x;
    const int row = blockIdx.y;
    const int bat = blockIdx.z;
    const __bf16* xb = x + (size_t)bat*HW*CC;

    for (int w = threadIdx.x; w < 3*34*8; w += 256) {   // 816 x 16B chunks
        int chunk = w & 7;
        int rc    = w >> 3;
        int r     = rc / 34;
        int col   = rc % 34;
        int gy = row + r - 1;
        int gx = bx*32 + col - 1;
        __bf16* dst = &xs[(r*34 + col)*CC + chunk*8];
        if ((unsigned)gy < HH && (unsigned)gx < WWD) {
            async_cp16(xb + ((size_t)gy*WWD + gx)*CC + chunk*8, dst);
        } else {
            v8bf z;
#pragma unroll
            for (int q = 0; q < 8; ++q) z[q] = (__bf16)0.0f;
            *(v8bf*)dst = z;
        }
    }
    async_wait();
    __syncthreads();

    const int wave = threadIdx.x >> 5;
    const int lane = threadIdx.x & 31;
    const int half = lane >> 4;
    const int idx  = lane & 15;
    const int mt   = wave >> 1;          // out-channel tile 0..3
    const int pt   = wave & 1;           // pixel tile 0..1

    v8f acc = {0.f,0.f,0.f,0.f,0.f,0.f,0.f,0.f};
#pragma unroll
    for (int t = 0; t < 9; ++t) {
        const int dy = t / 3, dx = t % 3;
#pragma unroll
        for (int kc = 0; kc < 2; ++kc) {
            const __bf16* ap = wA + ((t*CC + mt*16 + idx)*CC + kc*32 + half*8);
            union { v16bf v; v8bf h[2]; } A;
            A.h[0] = *(const v8bf*)(ap);
            A.h[1] = *(const v8bf*)(ap + 16);
            const __bf16* bp = &xs[(dy*34 + pt*16 + idx + dx)*CC + kc*32 + half*8];
            union { v16bf v; v8bf h[2]; } Bm;
            Bm.h[0] = *(const v8bf*)(bp);
            Bm.h[1] = *(const v8bf*)(bp + 16);
            acc = __builtin_amdgcn_wmma_f32_16x16x32_bf16(
                false, A.v, false, Bm.v, (short)0, acc, false, false);
        }
    }

    const float al = PRELU ? alpha[0] : 0.f;
    const int px  = bx*32 + pt*16 + idx;
    const int co0 = mt*16 + half*8;                     // C/D layout: M = r + 8*half
    if (RESID) {
#pragma unroll
        for (int r = 0; r < 8; ++r) {
            const int co = co0 + r;
            float v = acc[r] + bias[co];
            size_t off = (size_t)bat*CC*HW + (size_t)co*HW + (size_t)row*WWD + px;
            yf32[off] = v + resid[off];                 // NCHW f32 out (coalesced in px)
        }
    } else {
        v8bf ov;
#pragma unroll
        for (int r = 0; r < 8; ++r) {
            float v = acc[r] + bias[co0 + r];
            if (PRELU) v = (v >= 0.f) ? v : al*v;
            ov[r] = (__bf16)v;
        }
        *(v8bf*)(ybf + ((size_t)bat*HW + (size_t)row*WWD + px)*CC + co0) = ov;  // 16B store
    }
}

// --------------- fused 1x1 heads + target*guidance multiply ----------------
// kern written NHWC: [b][pix][72] f32 (two aligned float4 stores per lane).
__global__ __launch_bounds__(256)
void k_kg_head(const __bf16* __restrict__ ft, const __bf16* __restrict__ fg,
               const __bf16* __restrict__ wt, const __bf16* __restrict__ wg,
               const float* __restrict__ bt, const float* __restrict__ bg,
               float* __restrict__ kern) {
    __shared__ __bf16 fs[2*32*CC];       // [head][pix][ci]
    const int bx = blockIdx.x, row = blockIdx.y, bat = blockIdx.z;

    for (int w = threadIdx.x; w < 2*32*8; w += 256) {
        int chunk = w & 7;
        int rc    = w >> 3;
        int col   = rc & 31;
        int hd    = rc >> 5;
        const __bf16* src = (hd ? fg : ft) +
            ((size_t)bat*HW + (size_t)row*WWD + bx*32 + col)*CC + chunk*8;
        async_cp16(src, &fs[(hd*32 + col)*CC + chunk*8]);
    }
    async_wait();
    __syncthreads();

    const int wave = threadIdx.x >> 5;
    const int lane = threadIdx.x & 31;
    const int half = lane >> 4, idx = lane & 15;

    for (int it = wave; it < 10; it += 8) {     // 5 m-tiles x 2 p-tiles
        const int mt = it >> 1;
        const int pt = it & 1;
        v8f at = {0.f,0.f,0.f,0.f,0.f,0.f,0.f,0.f};
        v8f ag = {0.f,0.f,0.f,0.f,0.f,0.f,0.f,0.f};
#pragma unroll
        for (int kc = 0; kc < 2; ++kc) {
            union { v16bf v; v8bf h[2]; } A, Bm;
            const __bf16* bp0 = &fs[(pt*16 + idx)*CC       + kc*32 + half*8];
            const __bf16* bp1 = &fs[(32 + pt*16 + idx)*CC  + kc*32 + half*8];
            const __bf16* ap0 = wt + ((mt*16 + idx)*CC + kc*32 + half*8);
            A.h[0] = *(const v8bf*)(ap0); A.h[1] = *(const v8bf*)(ap0 + 16);
            Bm.h[0] = *(const v8bf*)(bp0); Bm.h[1] = *(const v8bf*)(bp0 + 16);
            at = __builtin_amdgcn_wmma_f32_16x16x32_bf16(
                false, A.v, false, Bm.v, (short)0, at, false, false);
            const __bf16* ap1 = wg + ((mt*16 + idx)*CC + kc*32 + half*8);
            A.h[0] = *(const v8bf*)(ap1); A.h[1] = *(const v8bf*)(ap1 + 16);
            Bm.h[0] = *(const v8bf*)(bp1); Bm.h[1] = *(const v8bf*)(bp1 + 16);
            ag = __builtin_amdgcn_wmma_f32_16x16x32_bf16(
                false, A.v, false, Bm.v, (short)0, ag, false, false);
        }
        if (mt == 4 && half == 1) continue;     // rows 72..79 are padding
        const int px = bx*32 + pt*16 + idx;
        const int m0 = mt*16 + half*8;
        v4f lo, hi;
#pragma unroll
        for (int r = 0; r < 4; ++r) {
            lo[r] = (at[r]     + bt[m0 + r])     * (ag[r]     + bg[m0 + r]);
            hi[r] = (at[r + 4] + bt[m0 + r + 4]) * (ag[r + 4] + bg[m0 + r + 4]);
        }
        float* kp = kern + ((size_t)bat*HW + (size_t)row*WWD + px)*KH + m0;
        *(v4f*)kp       = lo;
        *(v4f*)(kp + 4) = hi;
    }
}

// ------------- dynamic per-pixel filter (faithful raw-view sum) ------------
// out[c'] = sum_{t'} prod_flat[t'*64+c']; f = t'*64+c' -> g=f/72, cg=(f%72)/9,
// tap=f%9. All tensors NHWC (lane-consecutive channels -> coalesced).
__global__ __launch_bounds__(256)
void k_apply(const __bf16* __restrict__ imgn, const float* __restrict__ kern,
             __bf16* __restrict__ out) {
    int i = blockIdx.x * 256 + threadIdx.x;        // over BB*HW*CC, c fastest
    int c   = i & (CC-1);
    int p   = (i >> 6) % HW;
    int bat = (i >> 6) / HW;
    int y = p / WWD, x = p % WWD;
    const __bf16* ib = imgn + (size_t)bat*HW*CC;
    const float*  kb = kern + (size_t)bat*HW*KH + (size_t)p*KH;
    float s = 0.f;
#pragma unroll
    for (int tp = 0; tp < 9; ++tp) {
        int f   = tp*CC + c;
        int g   = f / KH;
        int rem = f % KH;
        int cg  = rem / 9;
        int tap = rem % 9;
        int ti = tap / 3, tj = tap % 3;
        int yy = y + ti - 1, xx = x + tj - 1;
        float pv = 0.f;
        if ((unsigned)yy < HH && (unsigned)xx < WWD)
            pv = (float)ib[((size_t)yy*WWD + xx)*CC + g*8 + cg];
        s += pv * kb[g*9 + tap];
    }
    out[i] = (__bf16)s;
}

// ------------------------------ launcher -----------------------------------

extern "C" void kernel_launch(void* const* d_in, const int* in_sizes, int n_in,
                              void* d_out, int out_size, void* d_ws, size_t ws_size,
                              hipStream_t stream) {
    (void)in_sizes; (void)n_in; (void)out_size; (void)ws_size;
    const float* image  = (const float*)d_in[0];
    const float* guid   = (const float*)d_in[1];
    const float* tkg_w  = (const float*)d_in[2];
    const float* tkg_b  = (const float*)d_in[3];
    const float* tkg_wk = (const float*)d_in[4];
    const float* tkg_bk = (const float*)d_in[5];
    const float* tkg_a  = (const float*)d_in[6];
    const float* gkg_w  = (const float*)d_in[7];
    const float* gkg_b  = (const float*)d_in[8];
    const float* gkg_wk = (const float*)d_in[9];
    const float* gkg_bk = (const float*)d_in[10];
    const float* gkg_a  = (const float*)d_in[11];
    const float* jbf_w  = (const float*)d_in[12];
    const float* jbf_b  = (const float*)d_in[13];
    const float* jbf_a  = (const float*)d_in[14];
    float* out = (float*)d_out;

    char* ws = (char*)d_ws;
    size_t off = 0;
    auto alloc = [&](size_t bytes) -> void* {
        void* p = ws + off;
        off = (off + bytes + 255) & ~(size_t)255;
        return p;
    };
    const size_t W3 = (size_t)9*CC*CC;
    __bf16* wt3 = (__bf16*)alloc(4*W3*2);
    __bf16* wg3 = (__bf16*)alloc(4*W3*2);
    __bf16* wj3 = (__bf16*)alloc(2*W3*2);
    __bf16* wht = (__bf16*)alloc((size_t)KHP*CC*2);
    __bf16* whg = (__bf16*)alloc((size_t)KHP*CC*2);
    const size_t ACT = (size_t)BB*HW*CC;                // NHWC bf16 activation elems
    __bf16* IMG = (__bf16*)alloc(ACT*2);                // NHWC image (persists)
    __bf16* A1  = (__bf16*)alloc(ACT*2);
    __bf16* B0  = (__bf16*)alloc(ACT*2);
    __bf16* B1  = (__bf16*)alloc(ACT*2);
    float*  KERNF = (float*)alloc((size_t)BB*HW*KH*4);  // NHWC per-pixel kernels

    // weight prep
    k_reorder_w3<<<(4*(int)W3 + 255)/256, 256, 0, stream>>>(tkg_w, wt3, 4*(int)W3);
    k_reorder_w3<<<(4*(int)W3 + 255)/256, 256, 0, stream>>>(gkg_w, wg3, 4*(int)W3);
    k_reorder_w3<<<(2*(int)W3 + 255)/256, 256, 0, stream>>>(jbf_w, wj3, 2*(int)W3);
    k_reorder_wh<<<(KHP*CC + 255)/256, 256, 0, stream>>>(tkg_wk, wht);
    k_reorder_wh<<<(KHP*CC + 255)/256, 256, 0, stream>>>(gkg_wk, whg);

    const int nact = BB*CC*HW;                          // 16,777,216
    k_nchw_to_nhwc<<<nact/256, 256, 0, stream>>>(image, IMG, nact);
    k_nchw_to_nhwc<<<nact/256, 256, 0, stream>>>(guid,  A1,  nact);

    dim3 cgrid(WWD/32, HH, BB);
    // target branch (final feat -> B1)
    k_conv3x3<true,false><<<cgrid,256,0,stream>>>(IMG, wt3 + 0*W3, tkg_b +   0, tkg_a, B0, nullptr, nullptr);
    k_conv3x3<true,false><<<cgrid,256,0,stream>>>(B0,  wt3 + 1*W3, tkg_b +  64, tkg_a, B1, nullptr, nullptr);
    k_conv3x3<true,false><<<cgrid,256,0,stream>>>(B1,  wt3 + 2*W3, tkg_b + 128, tkg_a, B0, nullptr, nullptr);
    k_conv3x3<true,false><<<cgrid,256,0,stream>>>(B0,  wt3 + 3*W3, tkg_b + 192, tkg_a, B1, nullptr, nullptr);
    // guidance branch (final feat -> A1)
    k_conv3x3<true,false><<<cgrid,256,0,stream>>>(A1,  wg3 + 0*W3, gkg_b +   0, gkg_a, B0, nullptr, nullptr);
    k_conv3x3<true,false><<<cgrid,256,0,stream>>>(B0,  wg3 + 1*W3, gkg_b +  64, gkg_a, A1, nullptr, nullptr);
    k_conv3x3<true,false><<<cgrid,256,0,stream>>>(A1,  wg3 + 2*W3, gkg_b + 128, gkg_a, B0, nullptr, nullptr);
    k_conv3x3<true,false><<<cgrid,256,0,stream>>>(B0,  wg3 + 3*W3, gkg_b + 192, gkg_a, A1, nullptr, nullptr);
    // fused heads + multiply -> per-pixel kernels (NHWC f32)
    k_kg_head<<<cgrid,256,0,stream>>>(B1, A1, wht, whg, tkg_bk, gkg_bk, KERNF);
    // dynamic filter -> NHWC bf16 features
    k_apply<<<nact/256, 256, 0, stream>>>(IMG, KERNF, B0);
    // jbf convs: conv+PReLU, then conv + residual (NCHW f32 out)
    k_conv3x3<true,false><<<cgrid,256,0,stream>>>(B0, wj3 + 0*W3, jbf_b +  0, jbf_a, A1, nullptr, nullptr);
    k_conv3x3<false,true><<<cgrid,256,0,stream>>>(A1, wj3 + 1*W3, jbf_b + 64, nullptr, nullptr, out, image);
}
